// BuildModel_3796751089773
// MI455X (gfx1250) — matrix operand order repaired and checked by reading, not compiled, not run on hardware
//
#include <hip/hip_runtime.h>

#define B_   1024
#define T_   256
#define F_   128
#define W_   3
#define H_   256
#define G4   1024
#define NTHR 512

typedef __bf16 v16bf __attribute__((ext_vector_type(16)));
typedef float  v8f   __attribute__((ext_vector_type(8)));
typedef float  v4f   __attribute__((ext_vector_type(4)));

__device__ __forceinline__ unsigned short f2bf(float f) {
  union { float f; unsigned int u; } v; v.f = f;
  unsigned int u = v.u;
  unsigned int r = (u + 0x7FFFu + ((u >> 16) & 1u)) >> 16;
  return (unsigned short)r;
}

// Branchless fast transcendentals (v_exp_f32 + v_rcp_f32, correct at +/-inf)
__device__ __forceinline__ float ftanh(float x) {
  float e = __expf(2.f * x);
  return 1.f - 2.f * __builtin_amdgcn_rcpf(e + 1.f);
}
__device__ __forceinline__ float fsig(float x) {
  return __builtin_amdgcn_rcpf(1.f + __expf(-x));
}

// ---------------------------------------------------------------------------
// Pack weights into bf16 WMMA B-matrix fragments.
// B-fragment (32x16, 16-bit): lanes 0-15 hold K=0..15 (col = lane), lanes
// 16-31 hold K=16..31 (col = lane-16); element j in lane = K (2 per VGPR).
// Fragment storage: [frag][lane*16 + j], 512 bf16 per 32x16 tile.
// Bm[k][n] = Wrow-major[n][k]  (computing  X @ W^T).
// ---------------------------------------------------------------------------
__global__ __launch_bounds__(256) void pack_weights(
    const float* __restrict__ w_hh, const float* __restrict__ w_ih,
    const float* __restrict__ w_h_beta,
    unsigned short* __restrict__ whh_p, unsigned short* __restrict__ wih_p,
    unsigned short* __restrict__ whb_p) {
  const int NWHH = 8 * 64 * 512;   // K=256 (8 kt), N=1024 (64 nt)
  const int NWIH = 4 * 64 * 512;   // K=128 (4 kt), N=1024 (64 nt)
  const int NWHB = 8 * 8 * 512;    // K=256 (8 kt), N=128  (8 nt)
  int idx = blockIdx.x * 256 + threadIdx.x;
  if (idx < NWHH) {
    int frag = idx >> 9, lj = idx & 511, ln = lj >> 4, j = lj & 15;
    int kt = frag >> 6, nt = frag & 63;
    int k = kt * 32 + j + 16 * (ln >> 4);
    int n = nt * 16 + (ln & 15);
    whh_p[idx] = f2bf(w_hh[n * H_ + k]);
  } else if (idx < NWHH + NWIH) {
    int e = idx - NWHH;
    int frag = e >> 9, lj = e & 511, ln = lj >> 4, j = lj & 15;
    int kt = frag >> 6, nt = frag & 63;
    int k = kt * 32 + j + 16 * (ln >> 4);
    int n = nt * 16 + (ln & 15);
    wih_p[e] = f2bf(w_ih[n * F_ + k]);
  } else if (idx < NWHH + NWIH + NWHB) {
    int e = idx - NWHH - NWIH;
    int frag = e >> 9, lj = e & 511, ln = lj >> 4, j = lj & 15;
    int kt = frag >> 3, nt = frag & 7;
    int k = kt * 32 + j + 16 * (ln >> 4);
    int n = nt * 16 + (ln & 15);
    whb_p[e] = f2bf(w_h_beta[n * H_ + k]);
  }
}

// ---------------------------------------------------------------------------
// Main recurrent kernel: one workgroup = 16 batch rows, all T steps.
// 512 threads = 16 waves; wave w owns gate N-tiles {w, w+16, w+32, w+48}
// (= i,f,g,o tiles of hidden-tile w); one c vector per wave in registers.
// A-fragment (16x32, 16-bit): lane m in 0-15 = row m, element j -> K:
//   k = 16*(lane/16) + j (same order as the B packing).
// ---------------------------------------------------------------------------
__global__ __launch_bounds__(NTHR) void raim_lstm_kernel(
    const float* __restrict__ v,
    const float* __restrict__ w_h_alpha, const float* __restrict__ b_h_alpha,
    const float* __restrict__ w_a_alpha, const float* __restrict__ b_a_alpha,
    const float* __restrict__ b_h_beta,
    const float* __restrict__ w_a_beta, const float* __restrict__ b_a_beta,
    const float* __restrict__ b_ih, const float* __restrict__ b_hh,
    const float* __restrict__ w_pred, const float* __restrict__ b_pred,
    const unsigned short* __restrict__ whh_p,
    const unsigned short* __restrict__ wih_p,
    const unsigned short* __restrict__ whb_p,
    float* __restrict__ out) {
  __shared__ __align__(16) float xsl[3][16][F_];        // rotating x window
  __shared__ __align__(16) float h_f32[16][H_];
  __shared__ __align__(16) float hsum[16][H_];
  __shared__ __align__(32) unsigned short h_frag[8][512];
  __shared__ __align__(32) unsigned short z_frag[4][512];
  __shared__ float beta_buf[16][F_];
  __shared__ float hx_buf[16][W_];
  __shared__ float alpha_buf[16][W_];
  __shared__ float bias_g[G4];
  __shared__ float s_bhb[F_], s_waa[F_], s_wpred[H_];
  __shared__ float s_wha[W_][H_];
  __shared__ float s_wab[W_], s_bha[W_];

  const int tid = threadIdx.x;
  const int lane = tid & 31, wave = tid >> 5;
  const int nl = lane & 15, gl = lane >> 4;
  const int b0 = blockIdx.x * 16;

  // ---- preamble: stage params, zero state ----
  for (int i = tid; i < G4; i += NTHR) bias_g[i] = b_ih[i] + b_hh[i];
  for (int i = tid; i < F_; i += NTHR) { s_bhb[i] = b_h_beta[i]; s_waa[i] = w_a_alpha[i]; }
  for (int i = tid; i < H_; i += NTHR) s_wpred[i] = w_pred[i];
  for (int i = tid; i < W_ * H_; i += NTHR) s_wha[i / H_][i % H_] = w_h_alpha[i];
  if (tid < W_) { s_wab[tid] = w_a_beta[tid]; s_bha[tid] = b_h_alpha[tid]; }
  for (int i = tid; i < 3 * 16 * F_; i += NTHR) ((float*)xsl)[i] = 0.f;
  for (int i = tid; i < 16 * H_; i += NTHR) { ((float*)h_f32)[i] = 0.f; ((float*)hsum)[i] = 0.f; }
  for (int i = tid; i < 8 * 512; i += NTHR) ((unsigned short*)h_frag)[i] = 0;
  const float ba_al = b_a_alpha[0], ba_be = b_a_beta[0];

  v8f creg;
  #pragma unroll
  for (int r = 0; r < 8; ++r) creg[r] = 0.f;

  __syncthreads();

  for (int t = 0; t < T_; ++t) {
    // Opaque *offset* (not pointer): LICM cannot hoist/CSE the weight-fragment
    // loads across t (address depends on a per-iteration black box), but the
    // base pointers remain kernarg-derived so address-space inference still
    // emits global_load (LOADcnt only) instead of flat_load (LOADcnt+DScnt).
    long toff = 0;
    asm volatile("" : "+s"(toff));
    const unsigned short* whh_t = whh_p + toff;
    const unsigned short* wih_t = wih_p + toff;
    const unsigned short* whb_t = whb_p + toff;

    const int slot = t % 3;
    // S1: stream v[:, t, :] into rotating slot (2048 f32 / 512 thr = 4 each)
    {
      int e = tid * 4;
      int m = e >> 7, f = e & 127;
      const v4f* src = (const v4f*)(v + ((size_t)(b0 + m) * T_ + t) * F_ + f);
      *(v4f*)(&xsl[slot][m][f]) = __builtin_nontemporal_load(src);
      if (t + 1 < T_)
        __builtin_prefetch(v + ((size_t)(b0 + m) * T_ + (t + 1)) * F_ + f, 0, 0);
    }
    __syncthreads();  // B1

    const int s0 = (t + 1) % 3, s1 = (t + 2) % 3, s2 = slot;

    // S2a: tid<48: alpha dot products (h.Wha^T + x.waa)
    if (tid < 48) {
      int w = tid % 3, row = tid / 3;
      int sw = (t + 1 + w) % 3;
      float acc = 0.f;
      #pragma unroll 4
      for (int k = 0; k < H_; ++k) acc += h_f32[row][k] * s_wha[w][k];
      #pragma unroll 4
      for (int f = 0; f < F_; ++f) acc += xsl[sw][row][f] * s_waa[f];
      hx_buf[row][w] = acc;
    }

    // S2b+S3: waves 0-7: hbeta tile `wave`, fuse xb + bias + tanh -> beta_buf
    if (wave < 8) {
      v8f accb;
      #pragma unroll
      for (int r = 0; r < 8; ++r) accb[r] = 0.f;
      #pragma unroll
      for (int kt = 0; kt < 8; ++kt) {
        v16bf a = *(const v16bf*)(&h_frag[kt][lane * 16]);
        v16bf bf = *(const v16bf*)(whb_t + ((kt * 8 + wave) * 512 + lane * 16));
        accb = __builtin_amdgcn_wmma_f32_16x16x32_bf16(false, a, false, bf,
                                                       (short)0, accb, false, false);
      }
      int col = wave * 16 + nl;
      #pragma unroll
      for (int r = 0; r < 8; ++r) {
        int m = r + 8 * gl;
        float xbv = s_wab[0] * xsl[s0][m][col] + s_wab[1] * xsl[s1][m][col] +
                    s_wab[2] * xsl[s2][m][col];
        beta_buf[m][col] = ftanh(accb[r] + s_bhb[col] + xbv + ba_be);
      }
    }
    __syncthreads();  // B3

    // S4: wave0 lanes 0-15: alpha softmax (3) + beta softmax (128) per row
    if (tid < 16) {
      int row = tid;
      float p0 = ftanh(hx_buf[row][0] + s_bha[0] + ba_al);
      float p1 = ftanh(hx_buf[row][1] + s_bha[1] + ba_al);
      float p2 = ftanh(hx_buf[row][2] + s_bha[2] + ba_al);
      float mx = fmaxf(p0, fmaxf(p1, p2));
      float e0 = __expf(p0 - mx), e1 = __expf(p1 - mx), e2 = __expf(p2 - mx);
      float inv = __builtin_amdgcn_rcpf(e0 + e1 + e2);
      alpha_buf[row][0] = e0 * inv; alpha_buf[row][1] = e1 * inv;
      alpha_buf[row][2] = e2 * inv;
      float bm = -1e30f;
      #pragma unroll 4
      for (int f = 0; f < F_; ++f) bm = fmaxf(bm, beta_buf[row][f]);
      float bs = 0.f;
      #pragma unroll 4
      for (int f = 0; f < F_; ++f) {
        float ex = __expf(beta_buf[row][f] - bm);
        beta_buf[row][f] = ex; bs += ex;
      }
      float binv = __builtin_amdgcn_rcpf(bs);
      #pragma unroll 4
      for (int f = 0; f < F_; ++f) beta_buf[row][f] *= binv;
    }
    __syncthreads();  // B4

    // S5: z = beta * sum_w alpha*x, packed straight into bf16 A-fragments
    {
      int e = tid * 4;
      #pragma unroll
      for (int q = 0; q < 4; ++q) {
        int idx = e + q;
        int kt = idx >> 9, lj = idx & 511, ln = lj >> 4, j = lj & 15;
        int m = ln & 15;
        int k = kt * 32 + 16 * (ln >> 4) + j;
        float xv = alpha_buf[m][0] * xsl[s0][m][k] +
                   alpha_buf[m][1] * xsl[s1][m][k] +
                   alpha_buf[m][2] * xsl[s2][m][k];
        z_frag[kt][lj] = f2bf(beta_buf[m][k] * xv);
      }
    }
    __syncthreads();  // B5

    // S6: gates = bias + z@Wih^T + h@Whh^T ; wave owns nt = wave + 16j
    v8f acc[4];
    {
      #pragma unroll
      for (int j = 0; j < 4; ++j) {
        float bj = bias_g[(wave + 16 * j) * 16 + nl];
        #pragma unroll
        for (int r = 0; r < 8; ++r) acc[j][r] = bj;
      }
    }
    #pragma unroll
    for (int kt = 0; kt < 4; ++kt) {
      v16bf a = *(const v16bf*)(&z_frag[kt][lane * 16]);
      #pragma unroll
      for (int j = 0; j < 4; ++j) {
        v16bf bf = *(const v16bf*)(wih_t + ((kt * 64 + wave + 16 * j) * 512 + lane * 16));
        acc[j] = __builtin_amdgcn_wmma_f32_16x16x32_bf16(false, a, false, bf,
                                                         (short)0, acc[j], false, false);
      }
    }
    #pragma unroll
    for (int kt = 0; kt < 8; ++kt) {
      v16bf a = *(const v16bf*)(&h_frag[kt][lane * 16]);
      #pragma unroll
      for (int j = 0; j < 4; ++j) {
        v16bf bf = *(const v16bf*)(whh_t + ((kt * 64 + wave + 16 * j) * 512 + lane * 16));
        acc[j] = __builtin_amdgcn_wmma_f32_16x16x32_bf16(false, a, false, bf,
                                                         (short)0, acc[j], false, false);
      }
    }

    // S7: LSTM nonlinearity; i,f,g,o = acc[0..3]; c in registers
    {
      #pragma unroll
      for (int r = 0; r < 8; ++r) {
        float ig = fsig(acc[0][r]);
        float fg = fsig(acc[1][r]);
        float gg = ftanh(acc[2][r]);
        float og = fsig(acc[3][r]);
        float cn = fg * creg[r] + ig * gg;
        creg[r] = cn;
        h_f32[r + 8 * gl][wave * 16 + nl] = og * ftanh(cn);
      }
    }
    __syncthreads();  // B6

    // S8: repack h -> bf16 A-fragments, accumulate hsum (bijective 16x256)
    {
      int e = tid * 8;
      #pragma unroll
      for (int q = 0; q < 8; ++q) {
        int idx = e + q;
        int kt = idx >> 9, lj = idx & 511, ln = lj >> 4, j = lj & 15;
        int m = ln & 15;
        int k = kt * 32 + 16 * (ln >> 4) + j;
        float hv = h_f32[m][k];
        hsum[m][k] += hv;
        h_frag[kt][lj] = f2bf(hv);
      }
    }
    // next-iteration B1 orders h_frag writes before their reads
  }

  __syncthreads();
  // head: out[b] = sum_t h_t . w_pred + T * b_pred
  if (tid < 16) {
    float acc = 0.f;
    #pragma unroll 4
    for (int k = 0; k < H_; ++k) acc += hsum[tid][k] * s_wpred[k];
    out[b0 + tid] = acc + (float)T_ * b_pred[0];
  }
}

extern "C" void kernel_launch(void* const* d_in, const int* in_sizes, int n_in,
                              void* d_out, int out_size, void* d_ws, size_t ws_size,
                              hipStream_t stream) {
  const float* v         = (const float*)d_in[0];
  const float* w_h_alpha = (const float*)d_in[1];
  const float* b_h_alpha = (const float*)d_in[2];
  const float* w_a_alpha = (const float*)d_in[3];
  const float* b_a_alpha = (const float*)d_in[4];
  const float* w_h_beta  = (const float*)d_in[5];
  const float* b_h_beta  = (const float*)d_in[6];
  const float* w_a_beta  = (const float*)d_in[7];
  const float* b_a_beta  = (const float*)d_in[8];
  const float* w_ih      = (const float*)d_in[9];
  const float* b_ih      = (const float*)d_in[10];
  const float* w_hh      = (const float*)d_in[11];
  const float* b_hh      = (const float*)d_in[12];
  const float* w_pred    = (const float*)d_in[13];
  const float* b_pred    = (const float*)d_in[14];

  unsigned short* whh_p = (unsigned short*)d_ws;             // 8*64*512 bf16
  unsigned short* wih_p = whh_p + 8 * 64 * 512;              // 4*64*512 bf16
  unsigned short* whb_p = wih_p + 4 * 64 * 512;              // 8*8*512  bf16

  const int total = 8 * 64 * 512 + 4 * 64 * 512 + 8 * 8 * 512;
  pack_weights<<<(total + 255) / 256, 256, 0, stream>>>(
      w_hh, w_ih, w_h_beta, whh_p, wih_p, whb_p);

  raim_lstm_kernel<<<B_ / 16, NTHR, 0, stream>>>(
      v, w_h_alpha, b_h_alpha, w_a_alpha, b_a_alpha,
      b_h_beta, w_a_beta, b_a_beta, b_ih, b_hh, w_pred, b_pred,
      whh_p, wih_p, whb_p, (float*)d_out);
}
